// MambaBlock_40063454937735
// MI455X (gfx1250) — matrix operand order repaired and checked by reading, not compiled, not run on hardware
//
#include <hip/hip_runtime.h>
#include <hip/hip_bf16.h>
#include <math.h>

#define D_MODEL  1024
#define D_INNER  2048
#define N_STATES 16
#define DT_RANK  64
#define KCONV    4
#define B_SZ     2
#define LSEQ     2048
#define XPLD     (2*D_INNER + N_STATES + DT_RANK)   // 4176
#define NTOK     (B_SZ*LSEQ)                        // 4096

typedef __attribute__((ext_vector_type(16))) __bf16 v16bf;
typedef __attribute__((ext_vector_type(8)))  __bf16 v8bf;
typedef __attribute__((ext_vector_type(4)))  __bf16 v4bf;
typedef __attribute__((ext_vector_type(8)))  float  v8f;

// ---------------------------------------------------------------------------
// gfx1250 async-to-LDS helpers (ASYNCcnt-tracked)
// ---------------------------------------------------------------------------
__device__ __forceinline__ void async_copy_b128(const __bf16* gsrc, unsigned lds_off) {
    // vdst = LDS byte address (addr[31:0] of a generic shared pointer), vaddr = 64-bit global
    asm volatile("global_load_async_to_lds_b128 %0, %1, off"
                 :: "v"(lds_off), "v"(gsrc) : "memory");
}
__device__ __forceinline__ void wait_async0() {
#if __has_builtin(__builtin_amdgcn_s_wait_asynccnt)
    __builtin_amdgcn_s_wait_asynccnt(0);
#else
    asm volatile("s_wait_asynccnt 0x0" ::: "memory");
#endif
}

// ---------------------------------------------------------------------------
// Fragment loaders (bf16), layouts per CDNA5 ISA 7.12.2
// ---------------------------------------------------------------------------

// A 16x32 bf16: lanes 0-15 -> M=0..15, K={0..7,16..23}; lanes 16-31 -> K={8..15,24..31}
__device__ __forceinline__ v16bf load_a_frag(const __bf16* __restrict__ A, int lda,
                                             int m0, int k0, int lane) {
    const int m  = m0 + (lane & 15);
    const int kb = (lane < 16) ? 0 : 8;
    const __bf16* row = A + (size_t)m * lda + k0 + kb;
    const v8bf lo = *reinterpret_cast<const v8bf*>(row);        // K = kb..kb+7
    const v8bf hi = *reinterpret_cast<const v8bf*>(row + 16);   // K = kb+16..kb+23
    return __builtin_shufflevector(lo, hi, 0,1,2,3,4,5,6,7,8,9,10,11,12,13,14,15);
}

// B 32x16 bf16 from an LDS tile stored [row(n)][k] (row stride 32 elements):
// lane = n; lanes 0-15 hold K={0..7,16..23}, lanes 16-31 hold K={8..15,24..31}
__device__ __forceinline__ v16bf load_b_lds(const __bf16* tile, int lane) {
    const int n  = lane & 15;
    const int kb = (lane < 16) ? 0 : 8;
    const __bf16* row = tile + n * 32 + kb;
    const v8bf lo = *reinterpret_cast<const v8bf*>(row);        // ds_load_b128
    const v8bf hi = *reinterpret_cast<const v8bf*>(row + 16);
    return __builtin_shufflevector(lo, hi, 0,1,2,3,4,5,6,7,8,9,10,11,12,13,14,15);
}

// ---------------------------------------------------------------------------
// WMMA GEMM: C[M,N] = A[M,K] @ W[N,K]^T + bias[N]   (EPI=1: softplus epilogue)
// 256 threads = 8 waves; wave -> one 16-row m-tile, block shares TN n-tiles.
// B-tiles staged to LDS with double-buffered async-to-LDS copies.
// ---------------------------------------------------------------------------
template<int TN, int EPI>
__global__ __launch_bounds__(256)
void gemm_wmma(const __bf16* __restrict__ A, int lda,
               const __bf16* __restrict__ W, int ldw,
               const float* __restrict__ bias,
               float* __restrict__ C, int ldc, int Kdim) {
    __shared__ __bf16 smem[2][TN * 16 * 32];   // [buf][(tile*16+n)*32 + k]

    const int lane = threadIdx.x & 31;
    const int wave = threadIdx.x >> 5;
    const int m0 = (blockIdx.x * 8 + wave) * 16;
    const int n0 = blockIdx.y * (TN * 16);

    // One 16B chunk per thread: row = global n-row within the TN*16 strip,
    // chunk = which 8-element span of the 32-wide k-slice.
    const int crow   = threadIdx.x >> 2;
    const int cchunk = threadIdx.x & 3;
    const bool do_copy = threadIdx.x < TN * 16 * 4;

    auto stage = [&](int buf, int k0) {
        if (do_copy) {
            const __bf16* g = W + (size_t)(n0 + crow) * ldw + k0 + cchunk * 8;
            const unsigned lds = (unsigned)(size_t)(&smem[buf][crow * 32 + cchunk * 8]);
            async_copy_b128(g, lds);
        }
    };

    stage(0, 0);

    v8f acc[TN] = {};
    int buf = 0;
    for (int k0 = 0; k0 < Kdim; k0 += 32, buf ^= 1) {
        wait_async0();
        __syncthreads();                       // buf is fully populated for all waves
        if (k0 + 32 < Kdim) stage(buf ^ 1, k0 + 32);

        const v16bf a = load_a_frag(A, lda, m0, k0, lane);
#pragma unroll
        for (int t = 0; t < TN; t++) {
            const v16bf b = load_b_lds(&smem[buf][t * 16 * 32], lane);
            acc[t] = __builtin_amdgcn_wmma_f32_16x16x32_bf16(
                false, a, false, b, (short)0, acc[t], false, false);
        }
    }

    // C/D: VGPR r -> M = r (+8 for high half-wave), N = lane&15
    const int ncol = lane & 15;
    const int mb   = m0 + ((lane < 16) ? 0 : 8);
#pragma unroll
    for (int t = 0; t < TN; t++) {
        const int n  = n0 + t * 16 + ncol;
        const float bv = bias[n];
#pragma unroll
        for (int r = 0; r < 8; r++) {
            float v = acc[t][r] + bv;
            if (EPI == 1) v = (v > 20.f) ? v : log1pf(__expf(v));  // softplus
            C[(size_t)(mb + r) * ldc + n] = v;
        }
    }
}

// ---------------------------------------------------------------------------
// f32 -> bf16 conversion (4 elements / thread)
// ---------------------------------------------------------------------------
__global__ __launch_bounds__(256)
void cvt_bf16_kernel(const float* __restrict__ src, __bf16* __restrict__ dst) {
    const size_t i = (size_t)blockIdx.x * blockDim.x + threadIdx.x;
    const float4 f = reinterpret_cast<const float4*>(src)[i];
    v4bf o; o[0] = (__bf16)f.x; o[1] = (__bf16)f.y; o[2] = (__bf16)f.z; o[3] = (__bf16)f.w;
    reinterpret_cast<v4bf*>(dst)[i] = o;
}

// Extract dr slice of xp (cols 4112..4176) as bf16: [NTOK, 64]
__global__ __launch_bounds__(256)
void dr_cvt_kernel(const float* __restrict__ xp, __bf16* __restrict__ drb) {
    const int idx = blockIdx.x * blockDim.x + threadIdx.x;   // [0, NTOK*16)
    const int r  = idx >> 4;
    const int c4 = (idx & 15) * 4;
    const float4 f = *reinterpret_cast<const float4*>(
        xp + (size_t)r * XPLD + 2 * D_INNER + N_STATES + c4);
    v4bf o; o[0] = (__bf16)f.x; o[1] = (__bf16)f.y; o[2] = (__bf16)f.z; o[3] = (__bf16)f.w;
    *reinterpret_cast<v4bf*>(drb + (size_t)r * DT_RANK + c4) = o;
}

// ---------------------------------------------------------------------------
// Depthwise causal conv (K=4) + SiLU on the x-slice of xp
// ---------------------------------------------------------------------------
__global__ __launch_bounds__(256)
void conv_silu_kernel(const float* __restrict__ xp,
                      const float* __restrict__ Wconv,
                      float* __restrict__ xconv) {
    const int idx = blockIdx.x * blockDim.x + threadIdx.x;   // [0, NTOK*D_INNER)
    const int d  = idx & (D_INNER - 1);
    const int bl = idx >> 11;
    const int b  = bl >> 11;
    const int l  = bl & (LSEQ - 1);
    float acc = 0.f;
#pragma unroll
    for (int t = 0; t < KCONV; t++) {
        const int ls = l - (KCONV - 1) + t;
        if (ls >= 0)
            acc += Wconv[d * KCONV + t] *
                   xp[(size_t)(b * LSEQ + ls) * XPLD + D_INNER + d];
    }
    xconv[idx] = acc * (1.f / (1.f + __expf(-acc)));  // silu
}

// Bmat = Bproj + B_param
__global__ __launch_bounds__(256)
void bmat_kernel(const float* __restrict__ xp,
                 const float* __restrict__ B_param,
                 float* __restrict__ bmat) {
    const int idx = blockIdx.x * blockDim.x + threadIdx.x;   // [0, NTOK*16)
    const int n  = idx & (N_STATES - 1);
    const int bl = idx >> 4;
    bmat[idx] = xp[(size_t)bl * XPLD + 2 * D_INNER + n] + B_param[n];
}

// ---------------------------------------------------------------------------
// Selective scan: one thread per (b,d) channel, N=16 state registers,
// sequential over L. Fuses y * silu(v); writes ys in bf16 (GEMM A-operand).
// ---------------------------------------------------------------------------
__global__ __launch_bounds__(256)
void scan_kernel(const float* __restrict__ xp,      // v slice: cols [0,D_INNER)
                 const float* __restrict__ xconv,   // [NTOK, D_INNER]
                 const float* __restrict__ delta,   // [NTOK, D_INNER]
                 const float* __restrict__ bmat,    // [NTOK, N_STATES]
                 const float* __restrict__ A_log,   // [D_INNER, N]
                 const float* __restrict__ Cp,      // [D_INNER, N]
                 __bf16* __restrict__ ys) {         // [NTOK, D_INNER] bf16
    const int tid = blockIdx.x * blockDim.x + threadIdx.x;   // 0..4095
    const int d = tid & (D_INNER - 1);
    const int b = tid >> 11;

    float Ad[N_STATES], Cd[N_STATES], h[N_STATES];
#pragma unroll
    for (int n = 0; n < N_STATES; n++) {
        Ad[n] = -__expf(A_log[(size_t)d * N_STATES + n]);
        Cd[n] = Cp[(size_t)d * N_STATES + n];
        h[n]  = 0.f;
    }

    size_t rowD = (size_t)b * LSEQ * D_INNER + d;
    size_t rowB = (size_t)b * LSEQ * N_STATES;
    size_t rowV = (size_t)b * LSEQ * XPLD + d;

    for (int l = 0; l < LSEQ; l++) {
        // hide L2 latency of the next timestep (global_prefetch_b8)
        __builtin_prefetch(delta + rowD + D_INNER, 0, 0);
        __builtin_prefetch(xconv + rowD + D_INNER, 0, 0);

        const float dv = delta[rowD];
        const float xv = xconv[rowD];
        const float vv = xp[rowV];
        const float dx = dv * xv;

        float y = 0.f;
#pragma unroll
        for (int n = 0; n < N_STATES; n++) {
            const float bn = bmat[rowB + n];
            h[n] = __expf(dv * Ad[n]) * h[n] + dx * bn;
            y += h[n] * Cd[n];
        }
        const float sv = vv * (1.f / (1.f + __expf(-vv)));
        ys[rowD] = (__bf16)(y * sv);

        rowD += D_INNER; rowB += N_STATES; rowV += XPLD;
    }
}

// ---------------------------------------------------------------------------
extern "C" void kernel_launch(void* const* d_in, const int* in_sizes, int n_in,
                              void* d_out, int out_size, void* d_ws, size_t ws_size,
                              hipStream_t stream) {
    const float* input   = (const float*)d_in[0];
    const float* Win     = (const float*)d_in[1];
    const float* b_in    = (const float*)d_in[2];
    const float* A_log   = (const float*)d_in[3];
    const float* B_param = (const float*)d_in[4];
    const float* C_param = (const float*)d_in[5];
    const float* Wdt     = (const float*)d_in[6];
    const float* b_dt    = (const float*)d_in[7];
    const float* Wconv   = (const float*)d_in[8];
    const float* Wout    = (const float*)d_in[9];
    const float* b_out   = (const float*)d_in[10];
    float* out = (float*)d_out;

    // ---- workspace layout (f32 region, then bf16 region) ----
    float* ws    = (float*)d_ws;
    size_t o = 0;
    float* xp    = ws + o;  o += (size_t)NTOK * XPLD;       // f32 [4096,4176]
    float* xconv = ws + o;  o += (size_t)NTOK * D_INNER;    // f32
    float* delta = ws + o;  o += (size_t)NTOK * D_INNER;    // f32
    float* bmat  = ws + o;  o += (size_t)NTOK * N_STATES;   // f32
    __bf16* bfw  = (__bf16*)(ws + o);
    size_t ob = 0;
    __bf16* in_b   = bfw + ob;  ob += (size_t)NTOK * D_MODEL;
    __bf16* win_b  = bfw + ob;  ob += (size_t)XPLD * D_MODEL;
    __bf16* wdt_b  = bfw + ob;  ob += (size_t)D_INNER * DT_RANK;
    __bf16* wout_b = bfw + ob;  ob += (size_t)D_MODEL * D_INNER;
    __bf16* dr_b   = bfw + ob;  ob += (size_t)NTOK * DT_RANK;
    __bf16* ys_b   = bfw + ob;  ob += (size_t)NTOK * D_INNER;

    const dim3 blk(256);

    // ---- one-time f32->bf16 operand conversions (bandwidth-trivial) ----
    cvt_bf16_kernel<<<dim3((NTOK * D_MODEL)   / 1024), blk, 0, stream>>>(input, in_b);
    cvt_bf16_kernel<<<dim3((XPLD * D_MODEL)   / 1024), blk, 0, stream>>>(Win,   win_b);
    cvt_bf16_kernel<<<dim3((D_INNER * DT_RANK)/ 1024), blk, 0, stream>>>(Wdt,   wdt_b);
    cvt_bf16_kernel<<<dim3((D_MODEL * D_INNER)/ 1024), blk, 0, stream>>>(Wout,  wout_b);

    // in_proj: [4096,1024] @ [1024,4176] + b_in ; N=4176 = 87 * (3*16)
    gemm_wmma<3, 0><<<dim3(32, 87), blk, 0, stream>>>(
        in_b, D_MODEL, win_b, D_MODEL, b_in, xp, XPLD, D_MODEL);

    conv_silu_kernel<<<dim3((NTOK * D_INNER) / 256), blk, 0, stream>>>(xp, Wconv, xconv);
    bmat_kernel<<<dim3((NTOK * N_STATES) / 256), blk, 0, stream>>>(xp, B_param, bmat);
    dr_cvt_kernel<<<dim3((NTOK * 16) / 256), blk, 0, stream>>>(xp, dr_b);

    // delta = softplus(dr @ Wdt^T + b_dt): K=64 ; N=2048 = 32 * (4*16)
    gemm_wmma<4, 1><<<dim3(32, 32), blk, 0, stream>>>(
        dr_b, DT_RANK, wdt_b, DT_RANK, b_dt, delta, D_INNER, DT_RANK);

    scan_kernel<<<dim3(16), blk, 0, stream>>>(xp, xconv, delta, bmat, A_log, C_param, ys_b);

    // out = ys @ Wout^T + b_out ; N=1024 = 16 * (4*16), K=2048
    gemm_wmma<4, 0><<<dim3(32, 16), blk, 0, stream>>>(
        ys_b, D_INNER, wout_b, D_INNER, b_out, out, D_MODEL, D_INNER);
}
